// DiffuserAttention_4380866641975
// MI455X (gfx1250) — matrix operand order repaired and checked by reading, not compiled, not run on hardware
//
#include <hip/hip_runtime.h>
#include <math.h>

// Problem constants (from reference)
#define NB   8192      // N = B*S nodes
#define DD   768       // hidden dim
#define HH   12        // heads
#define HDIM 64        // head dim
#define EE   131072    // edges
#define KDIM 768       // GEMM K

typedef __attribute__((ext_vector_type(16))) __bf16 v16bf;
typedef __attribute__((ext_vector_type(8)))  float  v8f;
typedef __attribute__((ext_vector_type(4)))  unsigned int u32x4;

union FragB { v16bf v; u32x4 q[2]; };

__device__ __forceinline__ unsigned short f32_to_bf16(float f) {
  unsigned u = __float_as_uint(f);
  u += 0x7FFFu + ((u >> 16) & 1u);   // round-to-nearest-even
  return (unsigned short)(u >> 16);
}

// ---------------------------------------------------------------- converts
__global__ void cvt_bf16_kernel(const float* __restrict__ in,
                                unsigned short* __restrict__ out, int n) {
  int i = blockIdx.x * blockDim.x + threadIdx.x;
  if (i < n) out[i] = f32_to_bf16(in[i]);
}

// ---------------------------------------------------------------- WMMA GEMM
// out[m][c] = (sum_k A[m][k]*W[c][k] + bias[c]) * scale (+ resid[m][c])
// A: [M x K] bf16 row-major, W: [Ncol x K] bf16 row-major (acts as B^T)
// block = 4 waves (2x2), each wave owns a 32x32 tile via 2x2 16x16x32 WMMAs.
__global__ __launch_bounds__(128)
void wmma_gemm_bf16_kernel(const unsigned short* __restrict__ A,
                           const unsigned short* __restrict__ W,
                           const float* __restrict__ bias,
                           const float* __restrict__ resid,
                           float* __restrict__ out,
                           float scale) {
  const int K = KDIM, Ncol = DD;
  const int lane = threadIdx.x & 31;
  const int wave = threadIdx.x >> 5;
  const int half = lane >> 4;     // 0/1 lane half
  const int ln   = lane & 15;
  const int row0 = blockIdx.x * 64 + (wave & 1) * 32;
  const int col0 = blockIdx.y * 64 + (wave >> 1) * 32;

  const unsigned short* pa0 = A + (size_t)(row0 + ln) * K;       // A tile i=0
  const unsigned short* pa1 = pa0 + (size_t)16 * K;              // A tile i=1
  const unsigned short* pb0 = W + (size_t)(col0 + ln) * K;       // B tile j=0
  const unsigned short* pb1 = pb0 + (size_t)16 * K;              // B tile j=1

  v8f c00 = {}, c01 = {}, c10 = {}, c11 = {};

  for (int k0 = 0; k0 < K; k0 += 32) {
    FragB a0, a1, b0, b1;
    // A fragment: element e<8 -> K=k0+8*half+e ; e>=8 -> K=k0+16+8*half+(e-8)
    const unsigned short* qa0 = pa0 + k0 + 8 * half;
    const unsigned short* qa1 = pa1 + k0 + 8 * half;
    a0.q[0] = *(const u32x4*)(qa0);
    a0.q[1] = *(const u32x4*)(qa0 + 16);
    a1.q[0] = *(const u32x4*)(qa1);
    a1.q[1] = *(const u32x4*)(qa1 + 16);
    // B fragment: element e<8 -> K=k0+8*half+e ; e>=8 -> K=k0+16+8*half+(e-8)
    const unsigned short* qb0 = pb0 + k0 + 8 * half;
    const unsigned short* qb1 = pb1 + k0 + 8 * half;
    b0.q[0] = *(const u32x4*)(qb0);
    b0.q[1] = *(const u32x4*)(qb0 + 16);
    b1.q[0] = *(const u32x4*)(qb1);
    b1.q[1] = *(const u32x4*)(qb1 + 16);

    // gfx1250 global_prefetch_b8 for the next K-slab (L2-resident stream)
    __builtin_prefetch((const void*)(qa0 + 64), 0, 1);
    __builtin_prefetch((const void*)(qb0 + 64), 0, 1);

    c00 = __builtin_amdgcn_wmma_f32_16x16x32_bf16(false, a0.v, false, b0.v,
                                                  (short)0, c00, false, false);
    c01 = __builtin_amdgcn_wmma_f32_16x16x32_bf16(false, a0.v, false, b1.v,
                                                  (short)0, c01, false, false);
    c10 = __builtin_amdgcn_wmma_f32_16x16x32_bf16(false, a1.v, false, b0.v,
                                                  (short)0, c10, false, false);
    c11 = __builtin_amdgcn_wmma_f32_16x16x32_bf16(false, a1.v, false, b1.v,
                                                  (short)0, c11, false, false);
  }

  // C/D layout: VGPR r holds M = r + 8*half, N = ln
  const int colA = col0 + ln;
  const int colB = col0 + 16 + ln;
  const float bA = bias[colA], bB = bias[colB];
#pragma unroll
  for (int r = 0; r < 8; ++r) {
    const int m0 = row0 + r + 8 * half;
    const int m1 = m0 + 16;
    float o00 = (c00[r] + bA) * scale;
    float o01 = (c01[r] + bB) * scale;
    float o10 = (c10[r] + bA) * scale;
    float o11 = (c11[r] + bB) * scale;
    if (resid) {
      o00 += resid[(size_t)m0 * Ncol + colA];
      o01 += resid[(size_t)m0 * Ncol + colB];
      o10 += resid[(size_t)m1 * Ncol + colA];
      o11 += resid[(size_t)m1 * Ncol + colB];
    }
    out[(size_t)m0 * Ncol + colA] = o00;
    out[(size_t)m0 * Ncol + colB] = o01;
    out[(size_t)m1 * Ncol + colA] = o10;
    out[(size_t)m1 * Ncol + colB] = o11;
  }
}

// ---------------------------------------------------------------- edge phase
__device__ __forceinline__ void atomicMaxFloat(float* addr, float val) {
  if (val >= 0.f) atomicMax((int*)addr, __float_as_int(val));
  else            atomicMin((unsigned int*)addr, __float_as_uint(val));
}

__global__ void fill_f32_kernel(float* __restrict__ p, float v, int n) {
  int i = blockIdx.x * blockDim.x + threadIdx.x;
  if (i < n) p[i] = v;
}

__global__ void edge_score_kernel(const float* __restrict__ q,
                                  const float* __restrict__ k,
                                  const int* __restrict__ esrc,
                                  const int* __restrict__ edst,
                                  const float* __restrict__ amask,
                                  float* __restrict__ score,
                                  float* __restrict__ smax) {
  int t = blockIdx.x * blockDim.x + threadIdx.x;
  if (t >= EE * HH) return;
  int e = t / HH, h = t - e * HH;
  int s = esrc[e], d = edst[e];
  const float4* kp = (const float4*)(k + (size_t)s * DD + h * HDIM);
  const float4* qp = (const float4*)(q + (size_t)d * DD + h * HDIM);
  float acc = 0.f;
#pragma unroll
  for (int i = 0; i < 16; ++i) {
    float4 kv = kp[i], qv = qp[i];
    acc += kv.x * qv.x + kv.y * qv.y + kv.z * qv.z + kv.w * qv.w;
  }
  if (!(amask[d] >= 0.f)) acc = -1e4f;
  score[t] = acc;
  atomicMaxFloat(smax + (size_t)d * HH + h, acc);
}

__global__ void pexp_kernel(float* __restrict__ score,
                            const float* __restrict__ smax,
                            float* __restrict__ denom,
                            const int* __restrict__ edst) {
  int t = blockIdx.x * blockDim.x + threadIdx.x;
  if (t >= EE * HH) return;
  int e = t / HH, h = t - e * HH;
  int d = edst[e];
  float p = expf(score[t] - smax[(size_t)d * HH + h]);
  score[t] = p;
  atomicAdd(denom + (size_t)d * HH + h, p);
}

__global__ void attn_norm_kernel(float* __restrict__ score,
                                 const float* __restrict__ denom,
                                 const int* __restrict__ edst) {
  int t = blockIdx.x * blockDim.x + threadIdx.x;
  if (t >= EE * HH) return;
  int e = t / HH, h = t - e * HH;
  int d = edst[e];
  score[t] = score[t] / denom[(size_t)d * HH + h];
}

// ---------------------------------------------------------------- propagation
__global__ void init_h_kernel(float* __restrict__ h, float* __restrict__ agg,
                              const float* __restrict__ v, int n) {
  int i = blockIdx.x * blockDim.x + threadIdx.x;
  if (i < n) { h[i] = v[i]; agg[i] = 0.f; }
}

__global__ void scatter_kernel(const float* __restrict__ hprev,
                               const float* __restrict__ attn,
                               const int* __restrict__ esrc,
                               const int* __restrict__ edst,
                               float* __restrict__ agg) {
  unsigned t = blockIdx.x * blockDim.x + threadIdx.x;   // EE*HH*16 threads
  unsigned e = t / (HH * 16);
  unsigned rem = t - e * (HH * 16);
  unsigned h = rem >> 4, g = rem & 15;
  float a = attn[(size_t)e * HH + h];
  int s = esrc[e], d = edst[e];
  const float4 hv = *(const float4*)(hprev + (size_t)s * DD + h * HDIM + g * 4);
  float* dp = agg + (size_t)d * DD + h * HDIM + g * 4;
  atomicAdd(dp + 0, hv.x * a);
  atomicAdd(dp + 1, hv.y * a);
  atomicAdd(dp + 2, hv.z * a);
  atomicAdd(dp + 3, hv.w * a);
}

__global__ void combine_kernel(float* __restrict__ h, float* __restrict__ agg,
                               const float* __restrict__ v, int n) {
  int i = blockIdx.x * blockDim.x + threadIdx.x;
  if (i < n) { h[i] = 0.9f * agg[i] + 0.1f * v[i]; agg[i] = 0.f; }
}

// ---------------------------------------------------------------- layernorm
__global__ __launch_bounds__(256)
void layernorm_kernel(const float* __restrict__ y, const float* __restrict__ g,
                      const float* __restrict__ b, float* __restrict__ out) {
  __shared__ float s1[256], s2[256];
  int row = blockIdx.x, tid = threadIdx.x;
  const float* yr = y + (size_t)row * DD;
  float a = 0.f, aa = 0.f;
  for (int i = tid; i < DD; i += 256) { float vv = yr[i]; a += vv; aa += vv * vv; }
  s1[tid] = a; s2[tid] = aa; __syncthreads();
  for (int o = 128; o > 0; o >>= 1) {
    if (tid < o) { s1[tid] += s1[tid + o]; s2[tid] += s2[tid + o]; }
    __syncthreads();
  }
  float mu  = s1[0] * (1.f / DD);
  float var = s2[0] * (1.f / DD) - mu * mu;
  float inv = rsqrtf(var + 1e-12f);
  for (int i = tid; i < DD; i += 256) {
    float vv = yr[i];
    out[(size_t)row * DD + i] = (vv - mu) * inv * g[i] + b[i];
  }
}

// ---------------------------------------------------------------- launcher
extern "C" void kernel_launch(void* const* d_in, const int* in_sizes, int n_in,
                              void* d_out, int out_size, void* d_ws, size_t ws_size,
                              hipStream_t stream) {
  const float* x    = (const float*)d_in[0];
  const float* am   = (const float*)d_in[1];
  const int*   esrc = (const int*)d_in[2];
  const int*   edst = (const int*)d_in[3];
  const float* Wq = (const float*)d_in[4];  const float* bq = (const float*)d_in[5];
  const float* Wk = (const float*)d_in[6];  const float* bk = (const float*)d_in[7];
  const float* Wv = (const float*)d_in[8];  const float* bv = (const float*)d_in[9];
  const float* Wo = (const float*)d_in[10]; const float* bo = (const float*)d_in[11];
  const float* lg = (const float*)d_in[12]; const float* lb = (const float*)d_in[13];
  float* out = (float*)d_out;

  const int ND = NB * DD;       // 6,291,456
  const int WW = DD * DD;       // 589,824
  const int EH = EE * HH;       // 1,572,864

  char* ws = (char*)d_ws;
  size_t off = 0;
  auto carve = [&](size_t bytes) -> char* {
    off = (off + 255) & ~(size_t)255;
    char* p = ws + off;
    off += bytes;
    return p;
  };
  unsigned short* xbf = (unsigned short*)carve((size_t)ND * 2);
  unsigned short* wqb = (unsigned short*)carve((size_t)WW * 2);
  unsigned short* wkb = (unsigned short*)carve((size_t)WW * 2);
  unsigned short* wvb = (unsigned short*)carve((size_t)WW * 2);
  unsigned short* wob = (unsigned short*)carve((size_t)WW * 2);
  float* qb    = (float*)carve((size_t)ND * 4);   // q, then reused as h
  float* kb    = (float*)carve((size_t)ND * 4);   // k, then reused as agg, then y
  float* vb    = (float*)carve((size_t)ND * 4);
  float* score = (float*)carve((size_t)EH * 4);   // score -> pexp -> attn (in place)
  float* smax  = (float*)carve((size_t)NB * HH * 4);
  float* denom = (float*)carve((size_t)NB * HH * 4);
  unsigned short* hbf = (unsigned short*)carve((size_t)ND * 2);

  dim3 blk(256);
  // 1) bf16 conversions (whole bf16 working set is L2-resident: ~17 MB of 192 MB)
  cvt_bf16_kernel<<<(ND + 255) / 256, blk, 0, stream>>>(x,  xbf, ND);
  cvt_bf16_kernel<<<(WW + 255) / 256, blk, 0, stream>>>(Wq, wqb, WW);
  cvt_bf16_kernel<<<(WW + 255) / 256, blk, 0, stream>>>(Wk, wkb, WW);
  cvt_bf16_kernel<<<(WW + 255) / 256, blk, 0, stream>>>(Wv, wvb, WW);
  cvt_bf16_kernel<<<(WW + 255) / 256, blk, 0, stream>>>(Wo, wob, WW);

  // 2) Q/K/V projections via WMMA (Q scaled by 1/sqrt(64))
  dim3 gg(NB / 64, DD / 64);
  wmma_gemm_bf16_kernel<<<gg, 128, 0, stream>>>(xbf, wqb, bq, nullptr, qb, 0.125f);
  wmma_gemm_bf16_kernel<<<gg, 128, 0, stream>>>(xbf, wkb, bk, nullptr, kb, 1.0f);
  wmma_gemm_bf16_kernel<<<gg, 128, 0, stream>>>(xbf, wvb, bv, nullptr, vb, 1.0f);

  // 3) edge scores + segment max/softmax
  fill_f32_kernel<<<(NB * HH + 255) / 256, blk, 0, stream>>>(smax, -3.0e38f, NB * HH);
  fill_f32_kernel<<<(NB * HH + 255) / 256, blk, 0, stream>>>(denom, 0.0f, NB * HH);
  edge_score_kernel<<<(EH + 255) / 256, blk, 0, stream>>>(qb, kb, esrc, edst, am, score, smax);
  pexp_kernel<<<(EH + 255) / 256, blk, 0, stream>>>(score, smax, denom, edst);
  attn_norm_kernel<<<(EH + 255) / 256, blk, 0, stream>>>(score, denom, edst);

  // 4) 5 propagation steps: h <- 0.9 * A h + 0.1 * v   (q buffer = h, k buffer = agg)
  init_h_kernel<<<(ND + 255) / 256, blk, 0, stream>>>(qb, kb, vb, ND);
  const unsigned scatterBlocks = (unsigned)(((size_t)EE * HH * 16) / 256);  // exact
  for (int s = 0; s < 5; ++s) {
    scatter_kernel<<<scatterBlocks, blk, 0, stream>>>(qb, score, esrc, edst, kb);
    combine_kernel<<<(ND + 255) / 256, blk, 0, stream>>>(qb, kb, vb, ND);
  }

  // 5) output projection (WMMA) with residual, then layernorm
  cvt_bf16_kernel<<<(ND + 255) / 256, blk, 0, stream>>>(qb, hbf, ND);
  float* yb = kb;  // agg buffer is free (zeroed) after last combine
  wmma_gemm_bf16_kernel<<<gg, 128, 0, stream>>>(hbf, wob, bo, x, yb, 1.0f);
  layernorm_kernel<<<NB, 256, 0, stream>>>(yb, lg, lb, out);

  (void)in_sizes; (void)n_in; (void)out_size; (void)ws_size;
}
